// Base_model_c_89859305767625
// MI455X (gfx1250) — compile-verified, matrix operand order checked by reading, not run on hardware
//
#include <hip/hip_runtime.h>
#include <hip/hip_bf16.h>
#include <math.h>

typedef __attribute__((ext_vector_type(16))) _Float16 v16h;
typedef __attribute__((ext_vector_type(8)))  _Float16 v8h;
typedef __attribute__((ext_vector_type(4)))  _Float16 v4h;
typedef __attribute__((ext_vector_type(8)))  float    v8f;

union AV { v16h v; v8h h8[2]; _Float16 h[16]; };
union CV { v8f  v; float f[8]; };

__device__ __forceinline__ float lrelu(float v) { return v >= 0.f ? v : 0.01f * v; }
__device__ __forceinline__ float sigm(float v)  { return 1.f / (1.f + __expf(-v)); }

// ---------------------------------------------------------------------------
// LDS-staged, register-pipelined WMMA GEMM:
//   C[M,Nout] = act(A[M,K] @ W[K,Nout] + bias)
//   - 256 threads = 8 waves (4m x 2n); block tile 128x64; wave tile 32x32
//     = 4 x V_WMMA_F32_16X16X32_F16 per K-step with A/B fragment reuse.
//   - K consumed 32 at a time through LDS (As[128][32] f16, Bs transposed [64][32]).
//   - Interior tiles: vectorized global_load_b128 fills; ragged edges (K=13/16,
//     M edge): clamped-address loads + select (branch-free, pipelined).
//   - Next K-step's tile is prefetched into registers while WMMAs run.
// ---------------------------------------------------------------------------
struct RegTile {
    float4 a[4];   // this thread's 16 A elements
    float4 b[2];   // this thread's 8 B elements
};

__device__ __forceinline__ void load_tile(RegTile& t,
    const float* __restrict__ A, int lda,
    const float* __restrict__ W, int Nout,
    int m0, int n0, int kk, int M, int K, int tid, bool aVec)
{
    const bool kFull = (kk + 32) <= K;
    // ---- A: 128x32 floats, 1024 float4 chunks, 4 per thread ----
    if (aVec && kFull && (m0 + 128) <= M) {
#pragma unroll
        for (int j = 0; j < 4; ++j) {
            int f4 = tid + j * 256;
            int r = f4 >> 3, c = (f4 & 7) << 2;
            t.a[j] = *(const float4*)(A + (size_t)(m0 + r) * lda + kk + c);
        }
    } else {
#pragma unroll
        for (int j = 0; j < 4; ++j) {
            int f4 = tid + j * 256;
            int r = f4 >> 3, c = (f4 & 7) << 2;
            int gm  = m0 + r;
            int gmc = gm < M ? gm : M - 1;
            bool mOk = gm < M;
            float vv[4];
#pragma unroll
            for (int q = 0; q < 4; ++q) {
                int gk  = kk + c + q;
                int gkc = gk < K ? gk : K - 1;
                float x = A[(size_t)gmc * lda + gkc];     // always in-bounds
                vv[q] = (mOk && gk < K) ? x : 0.f;        // select, no branch
            }
            t.a[j] = make_float4(vv[0], vv[1], vv[2], vv[3]);
        }
    }
    // ---- B: 32x64 floats, 512 float4 chunks, 2 per thread ----
    if (kFull) {
#pragma unroll
        for (int j = 0; j < 2; ++j) {
            int f4 = tid + j * 256;
            int k = f4 >> 4, nn = (f4 & 15) << 2;
            t.b[j] = *(const float4*)(W + (size_t)(kk + k) * Nout + n0 + nn);
        }
    } else {
#pragma unroll
        for (int j = 0; j < 2; ++j) {
            int f4 = tid + j * 256;
            int k = f4 >> 4, nn = (f4 & 15) << 2;
            int gk  = kk + k;
            int gkc = gk < K ? gk : K - 1;
            const float* row = W + (size_t)gkc * Nout + n0 + nn;
            float4 x = *(const float4*)(row);             // always in-bounds
            float z = (gk < K) ? 1.f : 0.f;
            t.b[j] = make_float4(x.x * z, x.y * z, x.z * z, x.w * z);
        }
    }
}

__device__ __forceinline__ void store_tile(const RegTile& t,
    _Float16* __restrict__ As, _Float16* __restrict__ Bs, int tid)
{
#pragma unroll
    for (int j = 0; j < 4; ++j) {
        int f4 = tid + j * 256;
        int r = f4 >> 3, c = (f4 & 7) << 2;
        v4h h = { (_Float16)t.a[j].x, (_Float16)t.a[j].y,
                  (_Float16)t.a[j].z, (_Float16)t.a[j].w };
        *(v4h*)(As + r * 32 + c) = h;                     // ds_store_b64
    }
#pragma unroll
    for (int j = 0; j < 2; ++j) {
        int f4 = tid + j * 256;
        int k = f4 >> 4, nn = (f4 & 15) << 2;
        Bs[(nn + 0) * 32 + k] = (_Float16)t.b[j].x;       // transposed [n][k]
        Bs[(nn + 1) * 32 + k] = (_Float16)t.b[j].y;
        Bs[(nn + 2) * 32 + k] = (_Float16)t.b[j].z;
        Bs[(nn + 3) * 32 + k] = (_Float16)t.b[j].w;
    }
}

__global__ void __launch_bounds__(256)
gemm_wmma_f16(const float* __restrict__ A, int lda,
              const float* __restrict__ W,
              const float* __restrict__ bias,
              float* __restrict__ C, int ldc,
              int M, int K, int Nout, int act)
{
    __shared__ __align__(16) _Float16 As[128 * 32];   // [m][k]
    __shared__ __align__(16) _Float16 Bs[64 * 32];    // [n][k] (transposed)

    const int tid    = threadIdx.x;
    const int lane   = tid & 31;
    const int wave   = tid >> 5;
    const int l16    = lane & 15;
    const int half   = lane >> 4;          // K-group select per ISA 7.12.2
    const int wave_m = wave & 3;           // 0..3
    const int wave_n = wave >> 2;          // 0..1
    const int m0     = blockIdx.x * 128;
    const int n0     = blockIdx.y * 64;
    const bool aVec  = ((lda & 3) == 0) && ((((size_t)A) & 15) == 0);

    CV acc[2][2];
#pragma unroll
    for (int mt = 0; mt < 2; ++mt)
#pragma unroll
        for (int nt = 0; nt < 2; ++nt)
#pragma unroll
            for (int r = 0; r < 8; ++r) acc[mt][nt].f[r] = 0.f;

    RegTile cur, nxt;
    load_tile(cur, A, lda, W, Nout, m0, n0, 0, M, K, tid, aVec);

    for (int kk = 0; kk < K; kk += 32) {
        __syncthreads();                   // LDS free (prev step consumed)
        store_tile(cur, As, Bs, tid);
        const bool more = (kk + 32) < K;
        if (more)                          // prefetch next tile during WMMAs
            load_tile(nxt, A, lda, W, Nout, m0, n0, kk + 32, M, K, tid, aVec);
        __syncthreads();

        AV a[2], b[2];
#pragma unroll
        for (int mt = 0; mt < 2; ++mt) {
            const _Float16* ap = As + (wave_m * 32 + mt * 16 + l16) * 32 + half * 8;
            a[mt].h8[0] = *(const v8h*)(ap);              // ds_load_b128
            a[mt].h8[1] = *(const v8h*)(ap + 16);
        }
#pragma unroll
        for (int nt = 0; nt < 2; ++nt) {
            const _Float16* bp = Bs + (wave_n * 32 + nt * 16 + l16) * 32 + half * 8;
            b[nt].h8[0] = *(const v8h*)(bp);
            b[nt].h8[1] = *(const v8h*)(bp + 16);
        }
#pragma unroll
        for (int mt = 0; mt < 2; ++mt)
#pragma unroll
            for (int nt = 0; nt < 2; ++nt)
                acc[mt][nt].v = __builtin_amdgcn_wmma_f32_16x16x32_f16(
                    false, a[mt].v, false, b[nt].v,
                    (short)0, acc[mt][nt].v, false, false);

        if (more) cur = nxt;
    }

    // --- epilogue: bias + leaky-relu, guarded M edge ---
#pragma unroll
    for (int nt = 0; nt < 2; ++nt) {
        int ncol = n0 + wave_n * 32 + nt * 16 + l16;
        float bv = bias ? bias[ncol] : 0.f;
#pragma unroll
        for (int mt = 0; mt < 2; ++mt) {
#pragma unroll
            for (int r = 0; r < 8; ++r) {
                int m = m0 + wave_m * 32 + mt * 16 + r + half * 8;
                if (m < M) {
                    float v = acc[mt][nt].f[r] + bv;
                    if (act) v = lrelu(v);
                    C[(size_t)m * ldc + ncol] = v;
                }
            }
        }
    }
}

// ---------------------------------------------------------------------------
// Graph kernels
// ---------------------------------------------------------------------------
__global__ void zero_kernel(float* __restrict__ p, long long cnt) {
    long long t = blockIdx.x * (long long)blockDim.x + threadIdx.x;
    if (t < cnt) p[t] = 0.f;
}

__global__ void deg_kernel(const int* __restrict__ dst, float* __restrict__ deg, int nE) {
    int e = blockIdx.x * blockDim.x + threadIdx.x;
    if (e < nE) atomicAdd(&deg[dst[e]], 1.0f);
}

__global__ void dinv_kernel(float* __restrict__ d, int n) {
    int i = blockIdx.x * blockDim.x + threadIdx.x;
    if (i < n) d[i] = rsqrtf(d[i] + 1.0f);    // self-loop: deg+1
}

// msg = hW[src] * dinv[src]; agg[dst] += msg   (float4 per thread, L2 atomics)
__global__ void scatter_kernel(const float* __restrict__ hW,
                               const int* __restrict__ src, const int* __restrict__ dst,
                               const float* __restrict__ dinv,
                               float* __restrict__ agg, int F, int nE)
{
    long long t = blockIdx.x * (long long)blockDim.x + threadIdx.x;
    int chunks = F >> 2;
    long long total = (long long)nE * chunks;
    if (t >= total) return;
    int e = (int)(t / chunks);
    int c = (int)(t % chunks);
    int s = src[e], d = dst[e];
    float sc = dinv[s];
    float4 v = ((const float4*)(hW + (size_t)s * F))[c];
    float* o = agg + (size_t)d * F + (size_t)c * 4;
    atomicAdd(o + 0, v.x * sc);
    atomicAdd(o + 1, v.y * sc);
    atomicAdd(o + 2, v.z * sc);
    atomicAdd(o + 3, v.w * sc);
}

// out = lrelu(dinv*agg + dinv^2*hW + b)
__global__ void combine_kernel(const float* __restrict__ hW, const float* __restrict__ agg,
                               const float* __restrict__ dinv, const float* __restrict__ bias,
                               float* __restrict__ out, int ldo, int F, int n)
{
    long long t = blockIdx.x * (long long)blockDim.x + threadIdx.x;
    long long total = (long long)n * F;
    if (t >= total) return;
    int i = (int)(t / F);
    int f = (int)(t % F);
    float di = dinv[i];
    float v = di * agg[t] + di * di * hW[t] + bias[f];
    out[(size_t)i * ldo + f] = lrelu(v);
}

__global__ void mul_inplace_kernel(float* __restrict__ a, const float* __restrict__ b, long long cnt) {
    long long t = blockIdx.x * (long long)blockDim.x + threadIdx.x;
    if (t < cnt) a[t] *= b[t];
}

// s = sigmoid( t[i,:64] . w + b )
__global__ void head64_kernel(const float* __restrict__ t, const float* __restrict__ w,
                              const float* __restrict__ b, float* __restrict__ out, int n)
{
    int i = blockIdx.x * blockDim.x + threadIdx.x;
    if (i >= n) return;
    const float* row = t + (size_t)i * 64;
    float s = b[0];
#pragma unroll 8
    for (int f = 0; f < 64; ++f) s += row[f] * w[f];
    out[i] = sigm(s);
}

// y = sigmoid( lrelu([sci,ssi] @ Wr1 + br1) @ Wr2 + br2 )
__global__ void final_kernel(const float* __restrict__ sci, const float* __restrict__ ssi,
                             const float* __restrict__ Wr1, const float* __restrict__ br1,
                             const float* __restrict__ Wr2, const float* __restrict__ br2,
                             float* __restrict__ y, int n)
{
    int i = blockIdx.x * blockDim.x + threadIdx.x;
    if (i >= n) return;
    float a0 = sci[i], a1 = ssi[i];
    float z0 = lrelu(a0 * Wr1[0] + a1 * Wr1[2] + br1[0]);
    float z1 = lrelu(a0 * Wr1[1] + a1 * Wr1[3] + br1[1]);
    y[i] = sigm(z0 * Wr2[0] + z1 * Wr2[1] + br2[0]);
}

// ---------------------------------------------------------------------------
// Host-side helpers
// ---------------------------------------------------------------------------
static void launch_gemm(const float* A, int lda, const float* W, const float* bias,
                        float* C, int ldc, int M, int K, int Nout, int act, hipStream_t s)
{
    dim3 grid((M + 127) / 128, (Nout + 63) / 64);
    gemm_wmma_f16<<<grid, 256, 0, s>>>(A, lda, W, bias, C, ldc, M, K, Nout, act);
}

static void launch_zero(float* p, long long cnt, hipStream_t s) {
    zero_kernel<<<(unsigned)((cnt + 255) / 256), 256, 0, s>>>(p, cnt);
}

// GCN layer: out = lrelu( dinv*segsum(hW[src]*dinv[src] -> dst) + dinv^2*hW + b )
static void launch_gcn(const float* h, int ldh, const float* W, const float* bias,
                       float* out, int ldo,
                       const int* src, const int* dst, const float* dinv,
                       float* hW, float* agg, int n, int K, int F, int nE, hipStream_t s)
{
    launch_gemm(h, ldh, W, nullptr, hW, F, n, K, F, /*act=*/0, s);
    launch_zero(agg, (long long)n * F, s);
    long long sc_threads = (long long)nE * (F >> 2);
    scatter_kernel<<<(unsigned)((sc_threads + 255) / 256), 256, 0, s>>>(hW, src, dst, dinv, agg, F, nE);
    long long cb = (long long)n * F;
    combine_kernel<<<(unsigned)((cb + 255) / 256), 256, 0, s>>>(hW, agg, dinv, bias, out, ldo, F, n);
}

extern "C" void kernel_launch(void* const* d_in, const int* in_sizes, int n_in,
                              void* d_out, int out_size, void* d_ws, size_t ws_size,
                              hipStream_t stream)
{
    const int n  = in_sizes[0] / 64;   // N_NODES
    const int nE = in_sizes[3];        // N_EDGES

    const float* discrete = (const float*)d_in[0];   // [n,64]
    const float* cont     = (const float*)d_in[1];   // [n,39]
    const float* churn    = (const float*)d_in[2];   // [n,16]
    const int*   eidx     = (const int*)d_in[4];     // [2,nE]
    const int*   src      = eidx;
    const int*   dst      = eidx + nE;

    const float *Wd  = (const float*)d_in[5],  *bd  = (const float*)d_in[6];
    const float *Wc1 = (const float*)d_in[7],  *bc1 = (const float*)d_in[8];
    const float *Wc2 = (const float*)d_in[9],  *bc2 = (const float*)d_in[10];
    const float *Wg0 = (const float*)d_in[11], *bg0 = (const float*)d_in[12];
    const float *Wg1 = (const float*)d_in[13], *bg1 = (const float*)d_in[14];
    const float *Wg2 = (const float*)d_in[15], *bg2 = (const float*)d_in[16];
    const float *Wfus= (const float*)d_in[17], *bfus= (const float*)d_in[18];
    const float *Wf0 = (const float*)d_in[19], *bf0 = (const float*)d_in[20];
    const float *Wf1 = (const float*)d_in[21], *bf1 = (const float*)d_in[22];
    const float *Wf2 = (const float*)d_in[23], *bf2 = (const float*)d_in[24];
    const float *Wns0= (const float*)d_in[25], *bns0= (const float*)d_in[26];
    const float *Wns1= (const float*)d_in[27], *bns1= (const float*)d_in[28];
    const float *Wl1 = (const float*)d_in[29], *bl1 = (const float*)d_in[30];
    const float *Wl2 = (const float*)d_in[31], *bl2 = (const float*)d_in[32];
    const float *Wl3 = (const float*)d_in[33], *bl3 = (const float*)d_in[34];
    const float *Wl4 = (const float*)d_in[35], *bl4 = (const float*)d_in[36];
    const float *Wr1 = (const float*)d_in[37], *br1 = (const float*)d_in[38];
    const float *Wr2 = (const float*)d_in[39], *br2 = (const float*)d_in[40];

    // d_out: y[n] | s_ci[n] | s_si[n] | h_ci[n*128] | h_si[n*128]
    float* out  = (float*)d_out;
    float* y    = out;
    float* s_ci = out + (size_t)n;
    float* s_si = out + (size_t)2 * n;
    float* h_ci = out + (size_t)3 * n;
    float* h_si = out + (size_t)3 * n + (size_t)n * 128;

    // workspace layout (floats)
    float* ws   = (float*)d_ws;
    float* dinv = ws;                                  // n
    float* Xcat = dinv + (size_t)n;                    // n*512 (x_d|x_c1|x_c2|x_g)
    float* bufA = Xcat + (size_t)n * 512;              // n*128 (hW / t)
    float* bufB = bufA + (size_t)n * 128;              // n*128 (agg)
    float* ns64 = bufB + (size_t)n * 128;              // n*64
    float* xns  = ns64 + (size_t)n * 64;               // n*128

    const unsigned nb  = (unsigned)((n + 255) / 256);
    const unsigned neb = (unsigned)((nE + 255) / 256);

    // 1) degree norm: dinv = (indeg+1)^-1/2
    launch_zero(dinv, n, stream);
    deg_kernel<<<neb, 256, 0, stream>>>(dst, dinv, nE);
    dinv_kernel<<<nb, 256, 0, stream>>>(dinv, n);

    // 2) input embeddings -> column slices of Xcat [n,512]
    launch_gemm(discrete,  64, Wd,  bd,  Xcat + 0,   512, n, 64, 128, 1, stream);
    launch_gemm(cont + 0,  39, Wc1, bc1, Xcat + 128, 512, n, 13, 128, 1, stream);
    launch_gemm(cont + 13, 39, Wc2, bc2, Xcat + 256, 512, n, 13, 128, 1, stream);
    launch_gemm(discrete,  64, Wg0, bg0, Xcat + 384, 512, n, 64, 128, 1, stream);

    // 3) two GCN layers on x_g (in place inside Xcat slice)
    launch_gcn(Xcat + 384, 512, Wg1, bg1, Xcat + 384, 512, src, dst, dinv, bufA, bufB, n, 128, 128, nE, stream);
    launch_gcn(Xcat + 384, 512, Wg2, bg2, Xcat + 384, 512, src, dst, dinv, bufA, bufB, n, 128, 128, nE, stream);

    // 4) fusion: h_ci = lrelu(Xcat @ Wfus + bfus)
    launch_gemm(Xcat, 512, Wfus, bfus, h_ci, 128, n, 512, 128, 1, stream);

    // 5) x_sf branch (staged in h_si slot of d_out)
    launch_gemm(discrete, 64, Wf0, bf0, h_si, 128, n, 64, 128, 1, stream);
    launch_gcn(h_si, 128, Wf1, bf1, h_si, 128, src, dst, dinv, bufA, bufB, n, 128, 128, nE, stream);
    launch_gcn(h_si, 128, Wf2, bf2, h_si, 128, src, dst, dinv, bufA, bufB, n, 128, 128, nE, stream);

    // 6) x_ns branch
    launch_gemm(churn, 16, Wns0, bns0, ns64, 64, n, 16, 64, 1, stream);
    launch_gcn(ns64, 64, Wns1, bns1, xns, 128, src, dst, dinv, bufA, bufB, n, 64, 128, nE, stream);

    // 7) h_si = x_sf * x_ns (in place in d_out)
    long long hcnt = (long long)n * 128;
    mul_inplace_kernel<<<(unsigned)((hcnt + 255) / 256), 256, 0, stream>>>(h_si, xns, hcnt);

    // 8) heads
    launch_gemm(h_ci, 128, Wl1, bl1, bufA, 64, n, 128, 64, 1, stream);
    head64_kernel<<<nb, 256, 0, stream>>>(bufA, Wl2, bl2, s_ci, n);
    launch_gemm(h_si, 128, Wl3, bl3, bufA, 64, n, 128, 64, 1, stream);
    head64_kernel<<<nb, 256, 0, stream>>>(bufA, Wl4, bl4, s_si, n);
    final_kernel<<<nb, 256, 0, stream>>>(s_ci, s_si, Wr1, br1, Wr2, br2, y, n);
}
